// MultiHeadAttention_56453050139535
// MI455X (gfx1250) — matrix operand order repaired and checked by reading, not compiled, not run on hardware
//
#include <hip/hip_runtime.h>

// ---------------- problem constants ----------------
#define BB 4
#define SS 2048
#define DD 1024
#define HH 16
#define DK 64
// M for activation GEMMs
#define MM (BB*SS)          // 8192

typedef __bf16 bf16_t;
typedef __attribute__((ext_vector_type(16))) __bf16 v16bf;
typedef __attribute__((ext_vector_type(8)))  __bf16 v8bf;
typedef __attribute__((ext_vector_type(8)))  float  v8f;

__device__ __forceinline__ v8f vzero8() {
    v8f z;
    #pragma unroll
    for (int i = 0; i < 8; ++i) z[i] = 0.0f;
    return z;
}
__device__ __forceinline__ v16bf vcombine(v8bf lo, v8bf hi) {
    v16bf a;
    #pragma unroll
    for (int i = 0; i < 8; ++i) { a[i] = lo[i]; a[i + 8] = hi[i]; }
    return a;
}
__device__ __forceinline__ v8f wmma_bf16(v16bf a, v16bf b, v8f c) {
    return __builtin_amdgcn_wmma_f32_16x16x32_bf16(false, a, false, b, (short)0, c, false, false);
}
// async copy 16 bytes global -> LDS (per lane), tracked by ASYNCcnt
__device__ __forceinline__ void async_b128(unsigned lds_off, const void* gaddr) {
    asm volatile("global_load_async_to_lds_b128 %0, %1, off"
                 :: "v"(lds_off), "v"((unsigned long long)gaddr) : "memory");
}
__device__ __forceinline__ void wait_async0() {
    asm volatile("s_wait_asynccnt 0" ::: "memory");
}

// ---------------- 1) fp32 -> bf16 elementwise ----------------
__global__ __launch_bounds__(256) void cvt_bf16_kernel(const float* __restrict__ x,
                                                       bf16_t* __restrict__ y, int n) {
    int i = (blockIdx.x * 256 + threadIdx.x) * 4;
    if (i + 3 < n) {
        float4 v = *(const float4*)(x + i);
        y[i + 0] = (bf16_t)v.x; y[i + 1] = (bf16_t)v.y;
        y[i + 2] = (bf16_t)v.z; y[i + 3] = (bf16_t)v.w;
    }
}

// ---------------- 2) transpose + convert weights: wT[n][k] = (bf16)w[k][n], 1024x1024 ----------------
__global__ __launch_bounds__(256) void tconv_kernel(const float* __restrict__ w,
                                                    bf16_t* __restrict__ wT) {
    __shared__ float tile[32][33];
    int bx = blockIdx.x * 32, by = blockIdx.y * 32;
    int tx = threadIdx.x, ty = threadIdx.y;           // 32 x 8
    #pragma unroll
    for (int i = 0; i < 32; i += 8)
        tile[ty + i][tx] = w[(size_t)(by + ty + i) * DD + bx + tx];
    __syncthreads();
    #pragma unroll
    for (int i = 0; i < 32; i += 8)
        wT[(size_t)(bx + ty + i) * DD + by + tx] = (bf16_t)tile[tx][ty + i];
}

// ---------------- 3) GEMM: C[M=8192, N=1024] = A[M,1024] * BT[N,1024]^T  (bf16, fp32 acc) -------
// Block: 8 waves, tile 128(M) x 64(N). Weight tiles staged in LDS via async loads, double buffered.
// MODE 0: store bf16 into heads layout  [b, h, s, d]   (Q, K projections)
// MODE 1: store bf16 into transposed    [b, h, d, s]   (V projection)
// MODE 3: store fp32 row-major [m, n]                   (final output projection)
template <int MODE>
__global__ __launch_bounds__(256) void gemm_kernel(const bf16_t* __restrict__ A,
                                                   const bf16_t* __restrict__ BT,
                                                   void* __restrict__ Cp) {
    __shared__ bf16_t bt[2][64 * 32];                 // [n_local][k_local]
    int tid = threadIdx.x, lane = tid & 31, wv = tid >> 5;
    int half = lane >> 4, ln = lane & 15;
    int n0 = blockIdx.x * 64;
    int m0 = blockIdx.y * 128 + wv * 16;

    // stage one 64x32 bf16 weight tile: 256 lanes x 16B
    int tn = tid >> 2, tc = tid & 3;
    const bf16_t* gsrc = BT + (size_t)(n0 + tn) * DD + tc * 8;
    unsigned ldst0 = (unsigned)(unsigned long long)&bt[0][tn * 32 + tc * 8];
    unsigned ldst1 = (unsigned)(unsigned long long)&bt[1][tn * 32 + tc * 8];

    async_b128(ldst0, gsrc);
    wait_async0();
    __syncthreads();

    v8f acc[4];
    #pragma unroll
    for (int t = 0; t < 4; ++t) acc[t] = vzero8();

    int buf = 0;
    for (int ks = 0; ks < DD; ks += 32) {
        if (ks + 32 < DD)
            async_b128(buf ? ldst0 : ldst1, gsrc + ks + 32);
        // A fragment: two contiguous 8-elem chunks per lane
        const bf16_t* ap = A + (size_t)(m0 + ln) * DD + ks + half * 16;
        v8bf alo = *(const v8bf*)ap;
        v8bf ahi = *(const v8bf*)(ap + 8);
        v16bf afrag = vcombine(alo, ahi);
        #pragma unroll
        for (int t = 0; t < 4; ++t) {
            v16bf bfrag = *(const v16bf*)&bt[buf][(t * 16 + ln) * 32 + half * 16];
            acc[t] = wmma_bf16(afrag, bfrag, acc[t]);
        }
        wait_async0();
        __syncthreads();
        buf ^= 1;
    }

    // store: C layout -> VGPR r: m = r + 8*half (lanes), n = ln
    #pragma unroll
    for (int t = 0; t < 4; ++t) {
        #pragma unroll
        for (int r = 0; r < 8; ++r) {
            int m = m0 + r + half * 8;
            int n = n0 + t * 16 + ln;
            float v = acc[t][r];
            if (MODE == 0) {
                int b = m >> 11, s = m & 2047, h = n >> 6, d = n & 63;
                ((bf16_t*)Cp)[(((size_t)(b * HH + h) * SS) + s) * DK + d] = (bf16_t)v;
            } else if (MODE == 1) {
                int b = m >> 11, s = m & 2047, h = n >> 6, d = n & 63;
                ((bf16_t*)Cp)[((size_t)(b * HH + h) * DK + d) * SS + s] = (bf16_t)v;
            } else {
                ((float*)Cp)[(size_t)m * DD + n] = v;
            }
        }
    }
}

// ---------------- 4) fused attention: scores -> softmax -> attn write -> P@V ----------------
// grid.x = B*H*(S/16) = 8192, block = 256 (8 waves). One block = 16 query rows, all 2048 keys.
__global__ __launch_bounds__(256) void attn_kernel(const bf16_t* __restrict__ qh,
                                                   const bf16_t* __restrict__ kh,
                                                   const bf16_t* __restrict__ vhT,
                                                   const unsigned char* __restrict__ mask,
                                                   float* __restrict__ attn,
                                                   bf16_t* __restrict__ oh) {
    __shared__ bf16_t q16[16 * DK];                  // 2 KB
    __shared__ float  sc[16 * SS];                   // 128 KB (CDNA5: 320 KB LDS per WGP)

    int blk = blockIdx.x;
    int i0 = (blk & 127) * 16;                       // query row block
    int bh = blk >> 7;                               // 0..63
    int b = bh >> 4, h = bh & 15;
    int tid = threadIdx.x, lane = tid & 31, wv = tid >> 5;
    int half = lane >> 4, ln = lane & 15;

    // ---- async-stage the 16x64 Q tile into LDS ----
    if (tid < 128) {
        unsigned lds = (unsigned)(unsigned long long)&q16[tid * 8];
        const bf16_t* g = qh + ((size_t)bh * SS + i0) * DK + tid * 8;
        async_b128(lds, g);
    }
    wait_async0();
    __syncthreads();

    // ---- scores: each wave does 256 key columns (16 WMMA tiles), K = 64 ----
    const bf16_t* Kbase = kh + (size_t)bh * SS * DK;
    v8f acc[16];
    #pragma unroll
    for (int t = 0; t < 16; ++t) acc[t] = vzero8();

    #pragma unroll
    for (int ks = 0; ks < DK; ks += 32) {
        int k0 = ks + half * 16;
        v8bf alo = *(const v8bf*)&q16[ln * DK + k0];
        v8bf ahi = *(const v8bf*)&q16[ln * DK + k0 + 8];
        v16bf afrag = vcombine(alo, ahi);
        #pragma unroll
        for (int t = 0; t < 16; ++t) {
            int col = wv * 256 + t * 16 + ln;        // key index
            v16bf bfrag = *(const v16bf*)(Kbase + (size_t)col * DK + ks + half * 16);
            acc[t] = wmma_bf16(afrag, bfrag, acc[t]);
        }
    }

    // ---- scale + mask, spill row-block of scores to LDS ----
    const float scale = 0.125f;                      // 1/sqrt(64)
    const unsigned char* mrow = mask + ((size_t)b * SS + i0) * SS;
    #pragma unroll
    for (int t = 0; t < 16; ++t) {
        #pragma unroll
        for (int r = 0; r < 8; ++r) {
            int m = r + half * 8;
            int col = wv * 256 + t * 16 + ln;
            float v = acc[t][r] * scale;
            v = mrow[(size_t)m * SS + col] ? v : -1e9f;
            sc[m * SS + col] = v;
        }
    }
    __syncthreads();

    // ---- softmax over 2048 columns; 16 lanes per row ----
    {
        int r = tid >> 4, cb = tid & 15;
        float* row = &sc[r * SS];
        float mx = -1e30f;
        for (int i = 0; i < 128; ++i) mx = fmaxf(mx, row[cb + i * 16]);
        #pragma unroll
        for (int o = 8; o > 0; o >>= 1) mx = fmaxf(mx, __shfl_xor(mx, o, 32));
        float sm = 0.0f;
        for (int i = 0; i < 128; ++i) {
            int c = cb + i * 16;
            float e = __expf(row[c] - mx);
            row[c] = e;
            sm += e;
        }
        #pragma unroll
        for (int o = 8; o > 0; o >>= 1) sm += __shfl_xor(sm, o, 32);
        float inv = 1.0f / sm;
        float* arow = attn + ((size_t)bh * SS + i0 + r) * SS;
        for (int i = 0; i < 128; ++i) {
            int c = cb + i * 16;
            float p = row[c] * inv;
            row[c] = p;
            arow[c] = p;                             // write attention probabilities
        }
    }
    __syncthreads();

    // ---- P @ V : out[16,64], K = 2048 split across 8 waves (256 each) ----
    const bf16_t* Vb = vhT + (size_t)bh * DK * SS;   // [d][s]
    v8f oacc[4];
    #pragma unroll
    for (int t = 0; t < 4; ++t) oacc[t] = vzero8();

    int kb0 = wv * 256;
    for (int ks = 0; ks < 256; ks += 32) {
        int k0 = kb0 + ks + half * 16;
        v16bf afrag;
        #pragma unroll
        for (int e = 0; e < 8; ++e) afrag[e] = (bf16_t)sc[ln * SS + k0 + e];
        #pragma unroll
        for (int e = 0; e < 8; ++e) afrag[e + 8] = (bf16_t)sc[ln * SS + k0 + 8 + e];
        #pragma unroll
        for (int t = 0; t < 4; ++t) {
            int d = t * 16 + ln;
            v16bf bfrag = *(const v16bf*)(Vb + (size_t)d * SS + kb0 + ks + half * 16);
            oacc[t] = wmma_bf16(afrag, bfrag, oacc[t]);
        }
    }
    __syncthreads();                                 // everyone done reading sc

    // ---- cross-wave reduction through LDS (reuse sc region) ----
    float* red = sc;                                 // 8*16*64 = 8192 floats
    #pragma unroll
    for (int t = 0; t < 4; ++t) {
        #pragma unroll
        for (int r = 0; r < 8; ++r) {
            int m = r + half * 8;
            red[(wv * 16 + m) * 64 + t * 16 + ln] = oacc[t][r];
        }
    }
    __syncthreads();
    for (int idx = tid; idx < 1024; idx += 256) {
        int m = idx >> 6, d = idx & 63;
        float s = 0.0f;
        #pragma unroll
        for (int w = 0; w < 8; ++w) s += red[(w * 16 + m) * 64 + d];
        oh[((size_t)b * SS + i0 + m) * DD + h * DK + d] = (bf16_t)s;
    }
}

// ---------------- host-side launcher ----------------
extern "C" void kernel_launch(void* const* d_in, const int* in_sizes, int n_in,
                              void* d_out, int out_size, void* d_ws, size_t ws_size,
                              hipStream_t stream) {
    const float* q    = (const float*)d_in[0];
    const float* k    = (const float*)d_in[1];
    const float* v    = (const float*)d_in[2];
    const unsigned char* mask = (const unsigned char*)d_in[3];
    const float* w_q  = (const float*)d_in[4];
    const float* w_k  = (const float*)d_in[5];
    const float* w_v  = (const float*)d_in[6];
    const float* w_fc = (const float*)d_in[7];

    char* ws = (char*)d_ws;
    const size_t MiB = 1024ull * 1024ull;
    bf16_t* qb   = (bf16_t*)(ws + 0 * MiB);     // [8192,1024] bf16
    bf16_t* kb   = (bf16_t*)(ws + 16 * MiB);
    bf16_t* vb   = (bf16_t*)(ws + 32 * MiB);
    bf16_t* wqT  = (bf16_t*)(ws + 48 * MiB);    // [n][k] bf16
    bf16_t* wkT  = (bf16_t*)(ws + 50 * MiB);
    bf16_t* wvT  = (bf16_t*)(ws + 52 * MiB);
    bf16_t* wfcT = (bf16_t*)(ws + 54 * MiB);
    bf16_t* qh   = (bf16_t*)(ws + 56 * MiB);    // [b,h,s,d]
    bf16_t* kh   = (bf16_t*)(ws + 72 * MiB);    // [b,h,s,d]
    bf16_t* vhT  = (bf16_t*)(ws + 88 * MiB);    // [b,h,d,s]
    bf16_t* oh   = (bf16_t*)(ws + 104 * MiB);   // [m,1024]

    float* out_p  = (float*)d_out;
    float* attn_p = out_p + (size_t)BB * SS * DD;   // out first, then attn

    const int nAct = MM * DD;                       // 8388608
    cvt_bf16_kernel<<<nAct / 1024, 256, 0, stream>>>(q, qb, nAct);
    cvt_bf16_kernel<<<nAct / 1024, 256, 0, stream>>>(k, kb, nAct);
    cvt_bf16_kernel<<<nAct / 1024, 256, 0, stream>>>(v, vb, nAct);

    dim3 tg(32, 32), tb(32, 8);
    tconv_kernel<<<tg, tb, 0, stream>>>(w_q, wqT);
    tconv_kernel<<<tg, tb, 0, stream>>>(w_k, wkT);
    tconv_kernel<<<tg, tb, 0, stream>>>(w_v, wvT);
    tconv_kernel<<<tg, tb, 0, stream>>>(w_fc, wfcT);

    dim3 gg(DD / 64, MM / 128);                     // (16, 64)
    gemm_kernel<0><<<gg, 256, 0, stream>>>(qb, wqT, (void*)qh);
    gemm_kernel<0><<<gg, 256, 0, stream>>>(kb, wkT, (void*)kh);
    gemm_kernel<1><<<gg, 256, 0, stream>>>(vb, wvT, (void*)vhT);

    attn_kernel<<<BB * HH * (SS / 16), 256, 0, stream>>>(qh, kh, vhT, mask, attn_p, oh);

    gemm_kernel<3><<<gg, 256, 0, stream>>>(oh, wfcT, (void*)d_out);
}
